// CrossAttention_979252544055
// MI455X (gfx1250) — matrix operand order repaired and checked by reading, not compiled, not run on hardware
//
#include <hip/hip_runtime.h>
#include <hip/hip_bf16.h>
#include <stdint.h>

// ---------------------------------------------------------------------------
// CrossAttention for MI455X (gfx1250, wave32, WMMA f32_16x16x32_f16 + TDM)
//   B=8, S=4096, C=77, D_EMBED=640, D_CONTEXT=768, H=8, DH=80
// Memory-bound (~340MB traffic vs ~61 GFLOP) -> f16 intermediates (resident
// in 192MB L2), WMMA for all big GEMMs, fused attention, TDM async loads.
// ---------------------------------------------------------------------------

typedef _Float16 half_t;
typedef __attribute__((ext_vector_type(16))) _Float16 v16h;
typedef __attribute__((ext_vector_type(8)))  _Float16 v8h;
typedef __attribute__((ext_vector_type(8)))  float    v8f;
typedef __attribute__((ext_vector_type(4)))  float    v4f;
typedef unsigned int u32x4 __attribute__((ext_vector_type(4)));
typedef int          i32x4 __attribute__((ext_vector_type(4)));
typedef int          i32x8 __attribute__((ext_vector_type(8)));

#define BB 8
#define SS 4096
#define CC 77
#define DE 640
#define DC 768
#define HH 8
#define DH 80
#define DHP 96    // d_head padded to 3*32 K-steps
#define CP 80     // context padded to 5*16 column tiles
#define DSTR 128  // ws row stride for K/V scratch (256B = TDM pad interval)
#define KVLD 136  // LDS row stride for K/V tiles (256B data + 16B pad)

#if defined(__has_builtin)
#  if __has_builtin(__builtin_amdgcn_tensor_load_to_lds)
#    define HAVE_TDM 1
#  else
#    define HAVE_TDM 0
#  endif
#else
#  define HAVE_TDM 0
#endif

// ---- WMMA fragment loaders (contiguous k = 16*hi + s, same order in A and B) ----------
__device__ __forceinline__ v16h frag_a(const half_t* row, int hi) {
  v8h c0 = *(const v8h*)(row + 16 * hi);
  v8h c1 = *(const v8h*)(row + 16 * hi + 8);
  v16h r;
#pragma unroll
  for (int j = 0; j < 8; ++j) { r[j] = c0[j]; r[j + 8] = c1[j]; }
  return r;
}
__device__ __forceinline__ v16h frag_b(const half_t* p) {
  v8h c0 = *(const v8h*)(p);
  v8h c1 = *(const v8h*)(p + 8);
  v16h r;
#pragma unroll
  for (int j = 0; j < 8; ++j) { r[j] = c0[j]; r[j + 8] = c1[j]; }
  return r;
}
__device__ __forceinline__ v8f wmma16(v16h a, v16h b, v8f c) {
  return __builtin_amdgcn_wmma_f32_16x16x32_f16(false, a, false, b,
                                                (short)0, c, false, false);
}

// ---- Tensor Data Mover: single-descriptor 2D tile load -> LDS -------------
// D# per ISA 8.3/8.4: g0 = {count|..., lds_addr, gaddr_lo, gaddr_hi|type=2},
// g1 = {mask|dsize|pad, tensor_dim0/1, tile_dim0/1/2, strides}.
#if HAVE_TDM
// flags: data_size=2B(code1), pad_enable, pad_interval, pad_amount
#define TDM_FLAGS_KV ((1u << 16) | (1u << 20) | (5u << 22) | (3u << 25)) // 64DW+4DW
#define TDM_FLAGS_A  ((1u << 16) | (1u << 20) | (2u << 22) | (3u << 25)) //  8DW+4DW
__device__ __forceinline__ void tdm_load(uint32_t lds_off, const void* gptr,
                                         uint32_t flags, uint32_t tile_d0,
                                         uint32_t tile_d1, uint32_t tens_d0,
                                         uint32_t tens_d1, uint64_t stride0) {
  const uint64_t ga = (uint64_t)(uintptr_t)gptr;
  u32x4 g0 = {0u, 0u, 0u, 0u};
  g0[0] = 1u;                                                // count = 1
  g0[1] = lds_off;                                           // lds_addr
  g0[2] = (uint32_t)ga;                                      // gaddr[31:0]
  g0[3] = (uint32_t)((ga >> 32) & 0x1FFFFFFull) | (2u << 30);// [56:32]|type=2
  i32x8 g1 = {0, 0, 0, 0, 0, 0, 0, 0};
  g1[0] = (int)flags;
  g1[1] = (int)((tens_d0 & 0xFFFFu) << 16);                  // dim0[15:0]
  g1[2] = (int)((tens_d0 >> 16) | ((tens_d1 & 0xFFFFu) << 16));
  g1[3] = (int)((tens_d1 >> 16) | (tile_d0 << 16));
  g1[4] = (int)(tile_d1 & 0xFFFFu);                          // tile_dim2 = 0
  g1[5] = (int)(stride0 & 0xFFFFFFFFull);                    // dim0 stride
  g1[6] = (int)((stride0 >> 32) & 0xFFFFull);
  g1[7] = 0;
  i32x4 z4 = {0, 0, 0, 0};
#if defined(__clang_major__) && (__clang_major__ >= 23)
  i32x8 z8 = {0, 0, 0, 0, 0, 0, 0, 0};
  __builtin_amdgcn_tensor_load_to_lds(g0, g1, z4, z4, z8, 0);
#else
  __builtin_amdgcn_tensor_load_to_lds(g0, g1, z4, z4, 0);
#endif
}
#endif  // HAVE_TDM

// ---------------------------------------------------------------------------
// Kernel 0: zero padded K/V scratch
// ---------------------------------------------------------------------------
__global__ void zero_ws(float4* p, int n) {
  int i = blockIdx.x * blockDim.x + threadIdx.x;
  if (i < n) p[i] = make_float4(0.f, 0.f, 0.f, 0.f);
}

// ---------------------------------------------------------------------------
// Kernel 1: K/V projection (tiny, ~1.2 GFLOP) with head split + padding.
//   Kp[b][h][c(80)][d(128)]  (c<77,d<80 valid, rest zero)
//   Vt[b][h][d(80)][c(128)]  (V transposed for the P@V B-operand)
// ---------------------------------------------------------------------------
__global__ __launch_bounds__(128) void kv_proj(
    const float* __restrict__ y, const float* __restrict__ Wk,
    const float* __restrict__ bk, const float* __restrict__ Wv,
    const float* __restrict__ bv, half_t* __restrict__ Kp,
    half_t* __restrict__ Vt) {
  const int b = blockIdx.z, c = blockIdx.y;
  const int n = blockIdx.x * 128 + threadIdx.x;  // 0..639
  const float* yr = y + ((size_t)b * CC + c) * DC;
  float ak = 0.f, av = 0.f;
  for (int k = 0; k < DC; ++k) {
    float yv = yr[k];
    ak = fmaf(yv, Wk[(size_t)k * DE + n], ak);
    av = fmaf(yv, Wv[(size_t)k * DE + n], av);
  }
  ak += bk[n];
  av += bv[n];
  const int h = n / DH, d = n % DH;
  Kp[((size_t)(b * HH + h) * CP + c) * DSTR + d] = (half_t)ak;
  Vt[((size_t)(b * HH + h) * DH + d) * DSTR + c] = (half_t)av;
}

// ---------------------------------------------------------------------------
// Kernels 2/4: WMMA GEMM out[M,N] = A[M,K] @ Bw[K,N] + bias (K=N=640).
// 128 threads (4 waves), 64x64 tile, BK=32. B transposed in LDS -> all
// fragment loads are contiguous ds_load_b128. f16-A tiles come in via TDM.
// ---------------------------------------------------------------------------
template <bool A_HALF, bool OUT_HALF>
__global__ __launch_bounds__(128) void gemm640(
    const void* __restrict__ A_, const float* __restrict__ Bw,
    const float* __restrict__ bias, void* __restrict__ out_,
    int M, int N, int K) {
  __shared__ alignas(16) half_t As[64][40];
  __shared__ alignas(16) half_t Bt[64][40];
  const int tid = threadIdx.x;
  const int lane = tid & 31, w = tid >> 5;
  const int hi = lane >> 4, m = lane & 15;
  const int m0 = blockIdx.y * 64, n0 = blockIdx.x * 64;

  const v8f vz = {0.f, 0.f, 0.f, 0.f, 0.f, 0.f, 0.f, 0.f};
  v8f acc[4] = {vz, vz, vz, vz};

  for (int k0 = 0; k0 < K; k0 += 32) {
    // --- A tile 64x32 -> LDS
    if constexpr (A_HALF) {
#if HAVE_TDM
      if (w == 0)  // one wave issues the DMA (TDM ignores EXEC; branch-guard)
        tdm_load((uint32_t)(uintptr_t)&As[0][0],
                 (const half_t*)A_ + (size_t)m0 * K + k0, TDM_FLAGS_A,
                 /*tile*/ 32u, 64u, /*tensor*/ 640u, 1u << 20, /*stride*/ 640ull);
#else
      const half_t* A = (const half_t*)A_;
#pragma unroll
      for (int i = 0; i < 2; ++i) {
        int idx = i * 128 + tid;
        int r = idx >> 2, c8 = idx & 3;
        *(v8h*)&As[r][8 * c8] =
            *(const v8h*)(A + (size_t)(m0 + r) * K + k0 + 8 * c8);
      }
#endif
    } else {
      const float* A = (const float*)A_;
#pragma unroll
      for (int i = 0; i < 4; ++i) {
        int idx = i * 128 + tid;
        int r = idx >> 3, c4 = idx & 7;
        v4f d = *(const v4f*)(A + (size_t)(m0 + r) * K + k0 + 4 * c4);
#pragma unroll
        for (int u = 0; u < 4; ++u) As[r][4 * c4 + u] = (half_t)d[u];
      }
    }
    // --- B tile 32x64 -> LDS transposed (N-major)
#pragma unroll
    for (int i = 0; i < 4; ++i) {
      int idx = i * 128 + tid;
      int kr = idx >> 4, c4 = idx & 15;
      v4f d = *(const v4f*)(Bw + (size_t)(k0 + kr) * N + n0 + 4 * c4);
#pragma unroll
      for (int u = 0; u < 4; ++u) Bt[4 * c4 + u][kr] = (half_t)d[u];
    }
    if (k0 + 32 < K) {  // speculative prefetch (global_prefetch_b8)
      if constexpr (!A_HALF)
        __builtin_prefetch((const float*)A_ + (size_t)(m0 + (tid & 63)) * K +
                               k0 + 32, 0, 0);
      __builtin_prefetch(Bw + (size_t)(k0 + 32 + (tid & 31)) * N + n0 +
                             ((tid >> 5) << 4), 0, 0);
    }
#if HAVE_TDM
    if constexpr (A_HALF) {
      if (w == 0) __builtin_amdgcn_s_wait_tensorcnt(0);  // TENSORcnt per-wave
    }
#endif
    __syncthreads();
    v16h a = frag_a(&As[w * 16 + m][0], hi);
#pragma unroll
    for (int j = 0; j < 4; ++j) {
      v16h b = frag_b(&Bt[j * 16 + m][16 * hi]);
      acc[j] = wmma16(a, b, acc[j]);
    }
    __syncthreads();
  }
#pragma unroll
  for (int j = 0; j < 4; ++j) {
    int col = n0 + j * 16 + m;
    float bvv = bias[col];
#pragma unroll
    for (int r = 0; r < 8; ++r) {
      int row = m0 + w * 16 + 8 * hi + r;
      float v = acc[j][r] + bvv;
      if constexpr (OUT_HALF)
        ((half_t*)out_)[(size_t)row * N + col] = (half_t)v;
      else
        ((float*)out_)[(size_t)row * N + col] = v;
    }
  }
}

// ---------------------------------------------------------------------------
// Kernel 3: fused attention per (b, h, 64 S-rows), 4 waves x 16-row strips.
// K/V tiles DMA'd by TDM (pad 256B->272B stride: conflict-free b128 frags).
// P reuses each wave's private q_s strip (dead after its score WMMAs).
// LDS: q 13312 + K 21760 + V 21760 = 56832 B.
// ---------------------------------------------------------------------------
__global__ __launch_bounds__(128) void attn64(
    const half_t* __restrict__ q, const half_t* __restrict__ Kp,
    const half_t* __restrict__ Vt, half_t* __restrict__ o) {
  __shared__ alignas(16) half_t q_s[64][104];   // [row][d 0..95] (pad zeroed)
  __shared__ alignas(16) half_t k_s[CP][KVLD];  // [c][d 0..127]
  __shared__ alignas(16) half_t v_s[DH][KVLD];  // [d][c 0..127]
  const int tid = threadIdx.x;
  const int lane = tid & 31, w = tid >> 5;
  const int hi = lane >> 4, m = lane & 15;
  const int b = blockIdx.z, h = blockIdx.y, m0 = blockIdx.x * 64;

  const half_t* kb = Kp + (size_t)(b * HH + h) * CP * DSTR;
  const half_t* vb = Vt + (size_t)(b * HH + h) * DH * DSTR;
#if HAVE_TDM
  if (w == 0) {  // 80*128 f16 = 20480B contiguous, LDS-padded every 256B
    tdm_load((uint32_t)(uintptr_t)&k_s[0][0], kb, TDM_FLAGS_KV,
             /*tile*/ 10240u, 1u, /*tensor*/ 10240u, 1u, /*stride*/ 10240ull);
    tdm_load((uint32_t)(uintptr_t)&v_s[0][0], vb, TDM_FLAGS_KV,
             10240u, 1u, 10240u, 1u, 10240ull);
  }
#else
#pragma unroll
  for (int i = 0; i < 10; ++i) {
    int idx = i * 128 + tid;  // 1280 chunks of 8 halves per matrix
    int r = idx >> 4, c8 = idx & 15;
    *(v8h*)&k_s[r][8 * c8] = *(const v8h*)(kb + (size_t)r * DSTR + 8 * c8);
    *(v8h*)&v_s[r][8 * c8] = *(const v8h*)(vb + (size_t)r * DSTR + 8 * c8);
  }
#endif

  // q tile: 64 rows x 80 halves, zero-pad d 80..95 (also serves as P pad)
#pragma unroll
  for (int i = 0; i < 5; ++i) {
    int idx = i * 128 + tid;  // 640 chunks of 8 halves
    int r = idx / 10, c8 = idx % 10;
    *(v8h*)&q_s[r][8 * c8] =
        *(const v8h*)(q + ((size_t)(b * SS + m0 + r) * DE + h * DH + 8 * c8));
  }
  {
    int r = tid >> 1, c = tid & 1;
    v8h z = {0, 0, 0, 0, 0, 0, 0, 0};
    *(v8h*)&q_s[r][DH + 8 * c] = z;
  }
#if HAVE_TDM
  if (w == 0) __builtin_amdgcn_s_wait_tensorcnt(0);
#endif
  __syncthreads();

  // ---- scores = q @ K^T (5 c-tiles x 3 K-steps)
  const v8f vz = {0.f, 0.f, 0.f, 0.f, 0.f, 0.f, 0.f, 0.f};
  v8f sacc[5] = {vz, vz, vz, vz, vz};
#pragma unroll
  for (int ks = 0; ks < DHP; ks += 32) {
    v16h a = frag_a(&q_s[w * 16 + m][ks], hi);
#pragma unroll
    for (int t = 0; t < 5; ++t)
      sacc[t] = wmma16(a, frag_b(&k_s[t * 16 + m][ks + 16 * hi]), sacc[t]);
  }

  // ---- masked softmax per row (16-lane shuffles stay inside half-wave)
  half_t(*p_row)[104] = (half_t(*)[104]) & q_s[w * 16][0];  // wave-private
  const float scale = 0.11180339887498948f;                 // 1/sqrt(80)
#pragma unroll
  for (int r = 0; r < 8; ++r) {
    float vals[5];
    float mx = -1e30f;
#pragma unroll
    for (int t = 0; t < 5; ++t) {
      float s = sacc[t][r] * scale;
      if (t * 16 + m >= CC) s = -1e30f;
      vals[t] = s;
      mx = fmaxf(mx, s);
    }
#pragma unroll
    for (int d = 1; d < 16; d <<= 1) mx = fmaxf(mx, __shfl_xor(mx, d, 32));
    float p[5];
    float sum = 0.f;
#pragma unroll
    for (int t = 0; t < 5; ++t) {
      p[t] = __expf(vals[t] - mx);
      sum += p[t];
    }
#pragma unroll
    for (int d = 1; d < 16; d <<= 1) sum += __shfl_xor(sum, d, 32);
    float inv = 1.0f / sum;
#pragma unroll
    for (int t = 0; t < 5; ++t)
      p_row[8 * hi + r][t * 16 + m] = (half_t)(p[t] * inv);
  }
  // wave-private LDS RAW (q strip -> P strip): enforce store->load order
  asm volatile("s_wait_dscnt 0x0" ::: "memory");

  // ---- out = P @ V (5 d-tiles x 3 K-steps)
  v8f oacc[5] = {vz, vz, vz, vz, vz};
#pragma unroll
  for (int cs = 0; cs < DHP; cs += 32) {
    v16h a = frag_a(&p_row[m][cs], hi);
#pragma unroll
    for (int t = 0; t < 5; ++t)
      oacc[t] = wmma16(a, frag_b(&v_s[t * 16 + m][cs + 16 * hi]), oacc[t]);
  }
#pragma unroll
  for (int t = 0; t < 5; ++t) {
#pragma unroll
    for (int r = 0; r < 8; ++r) {
      int row = m0 + w * 16 + 8 * hi + r;
      o[(size_t)(b * SS + row) * DE + h * DH + t * 16 + m] =
          (half_t)oacc[t][r];
    }
  }
}

// ---------------------------------------------------------------------------
// Host-side orchestration
// ---------------------------------------------------------------------------
extern "C" void kernel_launch(void* const* d_in, const int* in_sizes, int n_in,
                              void* d_out, int out_size, void* d_ws,
                              size_t ws_size, hipStream_t stream) {
  const float* x  = (const float*)d_in[0];
  const float* y  = (const float*)d_in[1];
  const float* Wq = (const float*)d_in[2];
  const float* bq = (const float*)d_in[3];
  const float* Wk = (const float*)d_in[4];
  const float* bk = (const float*)d_in[5];
  const float* Wv = (const float*)d_in[6];
  const float* bv = (const float*)d_in[7];
  const float* Wo = (const float*)d_in[8];
  const float* bo = (const float*)d_in[9];
  float* out = (float*)d_out;

  const size_t M = (size_t)BB * SS;  // 32768
  char* ws = (char*)d_ws;
  const size_t q_bytes  = M * DE * sizeof(half_t);  // 41.9 MB
  const size_t a_bytes  = M * DE * sizeof(half_t);  // 41.9 MB
  const size_t kp_bytes = (size_t)BB * HH * CP * DSTR * sizeof(half_t);
  half_t* qbuf = (half_t*)ws;
  half_t* abuf = (half_t*)(ws + q_bytes);
  half_t* Kp   = (half_t*)(ws + q_bytes + a_bytes);
  half_t* Vt   = (half_t*)(ws + q_bytes + a_bytes + kp_bytes);

  {  // 0) zero padded K/V scratch (2 * 1.31 MB)
    int n16 = (int)((2 * kp_bytes) / 16);
    zero_ws<<<(n16 + 255) / 256, 256, 0, stream>>>((float4*)Kp, n16);
  }
  // 1) K/V projection into padded, transposed-for-WMMA layouts
  kv_proj<<<dim3(5, CC, BB), 128, 0, stream>>>(y, Wk, bk, Wv, bv, Kp, Vt);
  // 2) Q projection: f32 x @ Wq + bq -> f16
  gemm640<false, true><<<dim3(DE / 64, (int)(M / 64)), 128, 0, stream>>>(
      (const void*)x, Wq, bq, (void*)qbuf, (int)M, DE, DE);
  // 3) fused attention -> f16 concat(head) layout
  attn64<<<dim3(SS / 64, HH, BB), 128, 0, stream>>>(qbuf, Kp, Vt, abuf);
  // 4) output projection: f16 attn @ Wo + bo -> f32
  gemm640<true, false><<<dim3(DE / 64, (int)(M / 64)), 128, 0, stream>>>(
      (const void*)abuf, Wo, bo, (void*)out, (int)M, DE, DE);
}
